// VectorQuantizer_4346506903728
// MI455X (gfx1250) — compile-verified, matrix operand order checked by reading, not run on hardware
//
#include <hip/hip_runtime.h>
#include <hip/hip_bf16.h>

// ---- CDNA5 WMMA vector types (probe-verified builtin signatures) ----
typedef __attribute__((ext_vector_type(16))) __bf16 v16bf;
typedef __attribute__((ext_vector_type(8)))  float  v8f;

#define K_CODES 512
#define DIM 64
#define HW  4096                  // 64*64
#define NBATCH 64
#define MROWS (NBATCH * HW)       // 262144 query vectors
#define OUT_ZQ_ELEMS (MROWS * DIM) // 16777216
#define LOSS_POS OUT_ZQ_ELEMS
#define IDX_POS  (OUT_ZQ_ELEMS + 1)

// workspace layout
#define WS_B_BYTES   (64 * 1024)                 // emb in bf16 WMMA-B layout
#define WS_ESQ_OFF   WS_B_BYTES                  // 512 floats ||e||^2
#define WS_PART_OFF  (WS_B_BYTES + K_CODES * 4)  // 65536 float loss partials
#define GATHER_BLOCKS (OUT_ZQ_ELEMS / 256)       // 65536

// ---------------------------------------------------------------------------
// Kernel 1: convert codebook (K=512, D=64) f32 -> bf16 in the exact per-lane
// layout of the WMMA 16-bit B matrix (32x16 per step), and compute e_sq[k].
// One (ktile, d-half, lane) triple per thread: 2048 threads total.
// B blob layout: [(kt*2 + half)*32 + lane] * 16 bf16 elements.
// Lane l: N = l&15 ; kb = (l>=16)?8:0 ; elem e -> d = half*32 + kb + (e<8 ? e : e+8)
// ---------------------------------------------------------------------------
__global__ void vq_prep(const float* __restrict__ emb,
                        __bf16* __restrict__ wsB,
                        float* __restrict__ wsEsq) {
    int tid = blockIdx.x * blockDim.x + threadIdx.x;   // 0..2047
    int kt = tid >> 6;
    int h  = (tid >> 5) & 1;
    int l  = tid & 31;
    int kb = (l >> 4) << 3;
    int k  = kt * 16 + (l & 15);
    __bf16* dst = wsB + (size_t)((kt * 2 + h) * 32 + l) * 16;
    const float* src = emb + k * DIM + h * 32 + kb;
#pragma unroll
    for (int e = 0; e < 16; ++e) {
        int d = (e < 8) ? e : (e + 8);                 // K pairs: 0..7 then +16
        dst[e] = (__bf16)src[d];
    }
    if (tid < K_CODES) {
        const float* row = emb + tid * DIM;
        float s = 0.0f;
#pragma unroll
        for (int d = 0; d < DIM; ++d) s = fmaf(row[d], row[d], s);
        wsEsq[tid] = s;
    }
}

// ---------------------------------------------------------------------------
// Kernel 2: argmin over K via WMMA bf16.
// One wave per 16-row tile of z (16384 tiles); 8 waves per 256-thread block.
// A-tile (16 x 64) gathered straight from the (N,D,H,W) tensor: for a tile
// inside batch n, element (m_local, d) sits at base + d*4096 + m_local
// (M contiguous, D strided) which matches per-lane dword gathers into the
// 16-bit A VGPR layout. The 64KB bf16 codebook is DMA'd into LDS once per
// block with the CDNA5 async global->LDS path (ASYNCcnt), no VGPR round-trip.
// ---------------------------------------------------------------------------
__global__ void __launch_bounds__(256) vq_argmin(
        const float* __restrict__ z_e,
        const __bf16* __restrict__ wsB,
        const float* __restrict__ wsEsq,
        float* __restrict__ idxOut) {
    __shared__ __align__(32) uint4 sB4[4096];          // 64 KB codebook in LDS

    // Async-DMA the codebook into LDS: 16 x b128 per thread, tracked by
    // ASYNCcnt. Generic pointers to LDS carry the LDS byte offset in their
    // low 32 bits (aperture mapping), which is what VDST wants.
    {
        const uint4* src = (const uint4*)wsB;
#pragma unroll 4
        for (int i = threadIdx.x; i < 4096; i += 256) {
            unsigned int      ldsOff = (unsigned int)(size_t)&sB4[i];
            unsigned long long gaddr = (unsigned long long)(size_t)(src + i);
            asm volatile("global_load_async_to_lds_b128 %0, %1, off"
                         :: "v"(ldsOff), "v"(gaddr) : "memory");
        }
        asm volatile("s_wait_asynccnt 0" ::: "memory");
    }
    __syncthreads();
    const v16bf* sB = (const v16bf*)sB4;

    int wave = threadIdx.x >> 5;
    int l    = threadIdx.x & 31;
    int tile = blockIdx.x * 8 + wave;                  // 0..16383
    int m0   = tile * 16;
    int n    = m0 >> 12;                               // / HW
    int tloc = m0 & 4095;
    int kb   = (l >> 4) << 3;

    // Load + convert the A tile (reused across all 32 k-tiles).
    const float* abase = z_e + (size_t)n * (DIM * HW) + tloc + (l & 15);
    v16bf aLo, aHi;
#pragma unroll
    for (int e = 0; e < 16; ++e) {
        int d = kb + ((e < 8) ? e : (e + 8));          // K in [0,32)
        aLo[e] = (__bf16)abase[(size_t)d * HW];
        aHi[e] = (__bf16)abase[(size_t)(d + 32) * HW];
    }

    float best[8];
    int   bidx[8];
#pragma unroll
    for (int r = 0; r < 8; ++r) { best[r] = 3.4e38f; bidx[r] = 0; }

    const float* esq = wsEsq + (l & 15);
    for (int kt = 0; kt < 32; ++kt) {
        v16bf bLo = sB[(kt * 2 + 0) * 32 + l];
        v16bf bHi = sB[(kt * 2 + 1) * 32 + l];
        v8f c = {};
        c = __builtin_amdgcn_wmma_f32_16x16x32_bf16(false, aLo, false, bLo,
                                                    (short)0, c, false, false);
        c = __builtin_amdgcn_wmma_f32_16x16x32_bf16(false, aHi, false, bHi,
                                                    (short)0, c, false, false);
        float es = esq[kt * 16];
        int   k  = kt * 16 + (l & 15);
        // distance (up to the row-constant ||z||^2): e_sq - 2*dot
#pragma unroll
        for (int r = 0; r < 8; ++r) {
            float dist = fmaf(-2.0f, c[r], es);
            if (dist < best[r]) { best[r] = dist; bidx[r] = k; }
        }
    }

    // C layout: VGPR r, lanes 0-15 -> row M=r ; lanes 16-31 -> row M=r+8.
    // Reduce argmin across the 16 lanes sharing each row (tie -> lower index).
#pragma unroll
    for (int off = 1; off < 16; off <<= 1) {
#pragma unroll
        for (int r = 0; r < 8; ++r) {
            float ov = __shfl_xor(best[r], off, 32);
            int   oi = __shfl_xor(bidx[r], off, 32);
            if (ov < best[r] || (ov == best[r] && oi < bidx[r])) {
                best[r] = ov; bidx[r] = oi;
            }
        }
    }
    if ((l & 15) == 0) {
        int rowBase = m0 + ((l >> 4) << 3);
#pragma unroll
        for (int r = 0; r < 8; ++r)
            idxOut[rowBase + r] = (float)bidx[r];      // exact for idx < 2^24
    }
}

// ---------------------------------------------------------------------------
// Kernel 3: gather z_q = emb[idx], write (N,D,H,W) output coalesced, and
// produce per-block loss partials (f32, 256 elements summed per block).
// ---------------------------------------------------------------------------
__global__ void __launch_bounds__(256) vq_gather(
        const float* __restrict__ z_e,
        const float* __restrict__ emb,
        const float* __restrict__ idxF,
        float* __restrict__ zq,
        float* __restrict__ partials) {
    int e   = blockIdx.x * 256 + threadIdx.x;          // element in (N,D,H,W)
    int n   = e >> 18;                                 // / (D*HW)
    int rem = e & (DIM * HW - 1);
    int d   = rem >> 12;                               // / HW
    int t   = rem & (HW - 1);
    int m   = (n << 12) | t;
    int idx = (int)idxF[m];
    float q = emb[idx * DIM + d];
    zq[e] = q;                                         // straight-through == z_q
    float diff = z_e[e] - q;
    float s = diff * diff;
#pragma unroll
    for (int off = 16; off >= 1; off >>= 1)
        s += __shfl_xor(s, off, 32);
    __shared__ float wsum[8];
    int lane = threadIdx.x & 31, wid = threadIdx.x >> 5;
    if (lane == 0) wsum[wid] = s;
    __syncthreads();
    if (threadIdx.x == 0) {
        float b = 0.0f;
#pragma unroll
        for (int i = 0; i < 8; ++i) b += wsum[i];
        partials[blockIdx.x] = b;
    }
}

// ---------------------------------------------------------------------------
// Kernel 4: final loss reduction in double, divide by M*D.
// ---------------------------------------------------------------------------
__global__ void __launch_bounds__(256) vq_loss_final(
        const float* __restrict__ partials, float* __restrict__ lossOut) {
    double acc = 0.0;
    for (int i = threadIdx.x; i < GATHER_BLOCKS; i += 256)
        acc += (double)partials[i];
#pragma unroll
    for (int off = 16; off >= 1; off >>= 1)
        acc += __shfl_xor(acc, off, 32);
    __shared__ double wacc[8];
    int lane = threadIdx.x & 31, wid = threadIdx.x >> 5;
    if (lane == 0) wacc[wid] = acc;
    __syncthreads();
    if (threadIdx.x == 0) {
        double tot = 0.0;
#pragma unroll
        for (int i = 0; i < 8; ++i) tot += wacc[i];
        lossOut[0] = (float)(tot / (double)OUT_ZQ_ELEMS);
    }
}

extern "C" void kernel_launch(void* const* d_in, const int* in_sizes, int n_in,
                              void* d_out, int out_size, void* d_ws, size_t ws_size,
                              hipStream_t stream) {
    const float* z_e = (const float*)d_in[0];   // (64, 64, 64, 64) f32
    const float* emb = (const float*)d_in[1];   // (512, 64) f32

    float* out    = (float*)d_out;
    float* zq     = out;                        // 16777216 f32
    float* loss   = out + LOSS_POS;             // 1 f32
    float* idxOut = out + IDX_POS;              // 262144 (indices as f32)

    char*   ws     = (char*)d_ws;
    __bf16* wsB    = (__bf16*)ws;
    float*  wsEsq  = (float*)(ws + WS_ESQ_OFF);
    float*  wsPart = (float*)(ws + WS_PART_OFF);

    vq_prep      <<<8,              256, 0, stream>>>(emb, wsB, wsEsq);
    vq_argmin    <<<MROWS / 16 / 8, 256, 0, stream>>>(z_e, wsB, wsEsq, idxOut);
    vq_gather    <<<GATHER_BLOCKS,  256, 0, stream>>>(z_e, emb, idxOut, zq, wsPart);
    vq_loss_final<<<1,              256, 0, stream>>>(wsPart, loss);
}
